// SurvivalGraphTransformer_43851616092286
// MI455X (gfx1250) — matrix operand order repaired and checked by reading, not compiled, not run on hardware
//
#include <hip/hip_runtime.h>
#include <cstdint>
#include <math.h>

// ---------------------------------------------------------------------------
// SurvivalGraphTransformer forward for MI455X (gfx1250, wave32, WMMA bf16).
//
// ~120 GFLOP, working set < 70 MB (L2-resident on 192 MB L2) -> compute
// bound -> all heavy math via v_wmma_f32_16x16x32_bf16. Attention is
// flash-style (online softmax); the graph bias is pre-padded/aligned bf16 so
// the softmax reads it with unconditional b128 loads (no exec-mask branches).
// ---------------------------------------------------------------------------

#define E_DIM   256
#define N_HEAD  8
#define DHEAD   32
#define FFN_D   1024
#define NLAY    4
#define NPROT   2048
#define NCLIN   32
#define NGEN    64
#define PEK     16
#define NBATCH  4
#define NSPEC   3
#define LLEN    (NPROT + NSPEC)      // 2051
#define MROW    (NBATCH * LLEN)      // 8204
#define BSTRIDE 2080                 // padded bias row stride (64B-aligned tiles)
#define LNEPS   1e-5f

typedef unsigned short u16b;
typedef __attribute__((ext_vector_type(16))) __bf16 v16bf;
typedef __attribute__((ext_vector_type(8)))  float  v8f;

union Frag { uint4 u[2]; v16bf v; };

__device__ inline u16b f2bf(float x) {
  union { float f; unsigned u; } v; v.f = x;
  unsigned r = v.u + 0x7FFFu + ((v.u >> 16) & 1u);   // round-to-nearest-even
  return (u16b)(r >> 16);
}
__device__ inline float bf2f(u16b x) {
  union { unsigned u; float f; } v; v.u = ((unsigned)x) << 16;
  return v.f;
}

// A-matrix fragment (16x32 bf16): lane r (M=r), half hh: element j holds
// K = 8*hh + (j<8 ? j : j+8)  -> two contiguous runs of 8 bf16 (16B each).
__device__ inline v16bf frag_run8(const u16b* row, int hh) {
  Frag f; const uint4* p = (const uint4*)row;
  f.u[0] = p[hh];
  f.u[1] = p[hh + 2];
  return f.v;
}
// B-matrix fragment (32x16 bf16) from an N-major staging buffer: lane r (N=r),
// half hh: element j holds K = 8*hh + (j<8 ? j : j+8) -> two runs of 8 bf16.
__device__ inline v16bf frag_run16(const u16b* row, int hh) {
  Frag f; const uint4* p = (const uint4*)row;
  f.u[0] = p[hh];
  f.u[1] = p[hh + 2];
  return f.v;
}

__device__ inline float block_sum256(float v, float* scr) {  // blockDim==256
  const int lane = threadIdx.x & 31, w = threadIdx.x >> 5;
#pragma unroll
  for (int o = 16; o; o >>= 1) v += __shfl_xor(v, o, 32);
  if (lane == 0) scr[w] = v;
  __syncthreads();
  float s = (threadIdx.x < 8) ? scr[threadIdx.x] : 0.f;
  if (w == 0) {
#pragma unroll
    for (int o = 4; o; o >>= 1) s += __shfl_xor(s, o, 32);
    if (lane == 0) scr[0] = s;
  }
  __syncthreads();
  float r = scr[0];
  __syncthreads();
  return r;
}

// ---------------------------------------------------------------------------
// Prep kernels
// ---------------------------------------------------------------------------
__global__ __launch_bounds__(256) void weight_transpose_bf16(
    const float* __restrict__ W, u16b* __restrict__ WT, int Kd, int Nd) {
  size_t idx = (size_t)blockIdx.x * 256 + threadIdx.x;
  if (idx >= (size_t)Kd * Nd) return;
  int n = (int)(idx / Kd), k = (int)(idx % Kd);
  WT[idx] = f2bf(W[(size_t)k * Nd + n]);
}

// Padded bf16 attention-bias matrix: Bp[q][k] (token coords, zeros on the
// NSPEC special rows/cols and in the stride padding).
__global__ __launch_bounds__(256) void build_bias_pad(
    const float* __restrict__ Kdiff, u16b* __restrict__ Bp) {
  size_t idx = (size_t)blockIdx.x * 256 + threadIdx.x;
  if (idx >= (size_t)LLEN * BSTRIDE) return;
  int q = (int)(idx / BSTRIDE), k = (int)(idx % BSTRIDE);
  u16b v = 0;
  if (q >= NSPEC && k >= NSPEC && k < LLEN)
    v = f2bf(Kdiff[(size_t)(q - NSPEC) * NPROT + (k - NSPEC)]);
  Bp[idx] = v;
}

// ---------------------------------------------------------------------------
// Token construction
// ---------------------------------------------------------------------------
__global__ __launch_bounds__(256) void build_prot_tokens(
    const float* __restrict__ protein, const float* __restrict__ PE,
    const float* __restrict__ vw, const float* __restrict__ vb,
    const float* __restrict__ pew, const float* __restrict__ peb,
    const float* __restrict__ pemb, float* __restrict__ tokens) {
  const int n = blockIdx.x, b = blockIdx.y, e = threadIdx.x;
  float pe = peb[e];
#pragma unroll
  for (int j = 0; j < PEK; ++j) pe += PE[n * PEK + j] * pew[j * E_DIM + e];
  float v = protein[b * NPROT + n] * vw[e] + vb[e] + pemb[(size_t)n * E_DIM + e] + pe;
  tokens[((size_t)b * LLEN + NSPEC + n) * E_DIM + e] = v;
}

__global__ __launch_bounds__(256) void build_special_tokens(
    const float* __restrict__ clinical, const float* __restrict__ genomic,
    const float* __restrict__ cw, const float* __restrict__ cb,
    const float* __restrict__ cg, const float* __restrict__ cbeta,
    const float* __restrict__ gw, const float* __restrict__ gb,
    const float* __restrict__ gg, const float* __restrict__ gbeta,
    const float* __restrict__ cls_tok, const float* __restrict__ clin_tok,
    const float* __restrict__ gen_tok, float* __restrict__ tokens) {
  __shared__ float scr[8];
  const int b = blockIdx.x, e = threadIdx.x;
  float pc = cb[e];
  for (int j = 0; j < NCLIN; ++j) pc += clinical[b * NCLIN + j] * cw[j * E_DIM + e];
  float s = block_sum256(pc, scr), s2 = block_sum256(pc * pc, scr);
  float mu = s / E_DIM, var = s2 / E_DIM - mu * mu;
  float yc = fmaxf((pc - mu) * rsqrtf(var + LNEPS) * cg[e] + cbeta[e], 0.f);
  float pg = gb[e];
  for (int j = 0; j < NGEN; ++j) pg += genomic[b * NGEN + j] * gw[j * E_DIM + e];
  s = block_sum256(pg, scr); s2 = block_sum256(pg * pg, scr);
  mu = s / E_DIM; var = s2 / E_DIM - mu * mu;
  float yg = fmaxf((pg - mu) * rsqrtf(var + LNEPS) * gg[e] + gbeta[e], 0.f);
  size_t rb = (size_t)b * LLEN * E_DIM;
  tokens[rb + 0 * E_DIM + e] = cls_tok[e];
  tokens[rb + 1 * E_DIM + e] = clin_tok[e] + yc;
  tokens[rb + 2 * E_DIM + e] = gen_tok[e] + yg;
}

// ---------------------------------------------------------------------------
// LayerNorm: fp32 tokens -> bf16 activation
// ---------------------------------------------------------------------------
__global__ __launch_bounds__(256) void ln_to_bf16(
    const float* __restrict__ X, const float* __restrict__ g,
    const float* __restrict__ bta, u16b* __restrict__ Y) {
  __shared__ float scr[8];
  const int row = blockIdx.x, e = threadIdx.x;
  float x = X[(size_t)row * E_DIM + e];
  float s = block_sum256(x, scr);
  float s2 = block_sum256(x * x, scr);
  float mu = s * (1.0f / E_DIM);
  float var = s2 * (1.0f / E_DIM) - mu * mu;
  Y[(size_t)row * E_DIM + e] = f2bf((x - mu) * rsqrtf(var + LNEPS) * g[e] + bta[e]);
}

// ---------------------------------------------------------------------------
// Tiled WMMA GEMM:  C[M,N] = epilogue(A[M,K] @ W[K,N] + bias)
//   A: bf16 row-major, WT: bf16 N-major (pre-transposed).
//   MODE 0: store bf16   MODE 1: exact GELU -> bf16   MODE 2: += into fp32 C
// Block = 128 threads (4 waves), tile 64x64, BK=32, 2x2 WMMA tiles per wave.
// 2-stage software pipeline: global loads for slab k+1 issue before the WMMA
// compute of slab k, so VMEM latency overlaps matrix math.
// ---------------------------------------------------------------------------
template <int MODE>
__global__ __launch_bounds__(128) void gemm_bf16_wmma(
    const u16b* __restrict__ A, const u16b* __restrict__ WT,
    const float* __restrict__ bias, void* __restrict__ Cout,
    int M, int N, int K) {
  __shared__ __align__(16) u16b As[64][32];
  __shared__ __align__(16) u16b Bs[64][32];
  const int tid = threadIdx.x;
  const int wave = tid >> 5, lane = tid & 31;
  const int r = lane & 15, hh = lane >> 4;
  const int m0 = blockIdx.y * 64, n0 = blockIdx.x * 64;
  const int wm = (wave >> 1) * 32, wn = (wave & 1) * 32;
  const int srow = tid >> 2, scv = (tid & 3) * 8;     // staging row / col
  v8f acc[2][2] = {};
  uint4 a_r[2], b_r[2];

  // prologue: stage slab 0 into registers
#pragma unroll
  for (int i = 0; i < 2; ++i) {
    int row = srow + 32 * i;
    int gr = m0 + row;
    a_r[i] = make_uint4(0u, 0u, 0u, 0u);
    if (gr < M) a_r[i] = *(const uint4*)(A + (size_t)gr * K + scv);
    b_r[i] = *(const uint4*)(WT + (size_t)(n0 + row) * K + scv);
  }

  for (int k0 = 0; k0 < K; k0 += 32) {
    __syncthreads();                        // previous slab fully consumed
#pragma unroll
    for (int i = 0; i < 2; ++i) {
      *(uint4*)&As[srow + 32 * i][scv] = a_r[i];
      *(uint4*)&Bs[srow + 32 * i][scv] = b_r[i];
    }
    __syncthreads();
    if (k0 + 32 < K) {                      // prefetch next slab into regs
      int k1 = k0 + 32;
#pragma unroll
      for (int i = 0; i < 2; ++i) {
        int row = srow + 32 * i;
        int gr = m0 + row;
        a_r[i] = make_uint4(0u, 0u, 0u, 0u);
        if (gr < M) a_r[i] = *(const uint4*)(A + (size_t)gr * K + k1 + scv);
        b_r[i] = *(const uint4*)(WT + (size_t)(n0 + row) * K + k1 + scv);
      }
    }
    v16bf a0 = frag_run8(&As[wm + r][0], hh);
    v16bf a1 = frag_run8(&As[wm + 16 + r][0], hh);
    v16bf b0 = frag_run16(&Bs[wn + r][0], hh);
    v16bf b1 = frag_run16(&Bs[wn + 16 + r][0], hh);
    acc[0][0] = __builtin_amdgcn_wmma_f32_16x16x32_bf16(false, a0, false, b0, (short)0, acc[0][0], false, false);
    acc[0][1] = __builtin_amdgcn_wmma_f32_16x16x32_bf16(false, a0, false, b1, (short)0, acc[0][1], false, false);
    acc[1][0] = __builtin_amdgcn_wmma_f32_16x16x32_bf16(false, a1, false, b0, (short)0, acc[1][0], false, false);
    acc[1][1] = __builtin_amdgcn_wmma_f32_16x16x32_bf16(false, a1, false, b1, (short)0, acc[1][1], false, false);
  }
#pragma unroll
  for (int mi = 0; mi < 2; ++mi)
#pragma unroll
    for (int ni = 0; ni < 2; ++ni)
#pragma unroll
      for (int i = 0; i < 8; ++i) {
        int row = m0 + wm + 16 * mi + 8 * hh + i;  // C/D layout: VGPR i, half hh
        int col = n0 + wn + 16 * ni + r;
        if (row < M) {
          float v = acc[mi][ni][i] + bias[col];
          if constexpr (MODE == 0) {
            ((u16b*)Cout)[(size_t)row * N + col] = f2bf(v);
          } else if constexpr (MODE == 1) {
            v = 0.5f * v * (1.0f + erff(v * 0.70710678118654752440f));
            ((u16b*)Cout)[(size_t)row * N + col] = f2bf(v);
          } else {
            ((float*)Cout)[(size_t)row * N + col] += v;     // residual
          }
        }
      }
}

// ---------------------------------------------------------------------------
// Flash attention with graph bias.
// Grid (ceil(L/128), H, B), 128 threads = 4 waves, each wave owns 32 q rows.
// Per key-tile (32 keys): S = Q@K^T (4 WMMA) -> LDS -> online softmax per
// row-lane (+ padded bf16 bias via unconditional b128 loads) -> O += P@V.
// ---------------------------------------------------------------------------
__global__ __launch_bounds__(128) void attn_flash_wmma(
    const u16b* __restrict__ Q, const u16b* __restrict__ Kmat,
    const u16b* __restrict__ Vmat, const u16b* __restrict__ Bp,
    const float* __restrict__ bias_scale, u16b* __restrict__ Out) {
  const int b = blockIdx.z, h = blockIdx.y;
  const int tid = threadIdx.x;
  const int wave = tid >> 5, lane = tid & 31;
  const int r = lane & 15, hh = lane >> 4;
  const int qb = blockIdx.x * 128 + wave * 32;
  const size_t base = (size_t)b * LLEN;
  const int c0 = h * DHEAD;

  __shared__ __align__(16) u16b Kt[32][32];          // [key][dh]
  __shared__ __align__(16) u16b Vt[32][32];          // [dh][key] (transposed)
  __shared__ __align__(16) u16b Qs[4][32][32];       // per wave [q][dh]
  __shared__ float Sls[4][32][33];                   // padded, per wave
  __shared__ __align__(16) u16b Pls[4][32][32];      // per wave [q][key]
  __shared__ float rowv[4][32];

  { // stage own wave's Q tile, lane = row
    int q = qb + lane;
    uint4* dst = (uint4*)&Qs[wave][lane][0];
    if (q < LLEN) {
      const uint4* src = (const uint4*)(Q + (base + q) * E_DIM + c0);
#pragma unroll
      for (int i = 0; i < 4; ++i) dst[i] = src[i];
    } else {
      uint4 z = make_uint4(0u, 0u, 0u, 0u);
#pragma unroll
      for (int i = 0; i < 4; ++i) dst[i] = z;
    }
  }

  float m_i = -INFINITY, l_i = 0.f;
  v8f o[2][2] = {};
  const float scl = 0.17677669529663688f;            // 1/sqrt(d_head)
  const float bsc = bias_scale[h];
  const int nkt = (LLEN + 31) / 32;
  const int qrow = (qb + lane < LLEN) ? qb + lane : 0;  // clamped bias row
  const u16b* brow = Bp + (size_t)qrow * BSTRIDE;

  for (int kt = 0; kt < nkt; ++kt) {
    const int kbase = kt * 32;
    __syncthreads();
    { // stage K tile + V tile transposed (vectorized global reads)
      int row = tid >> 2, cv = (tid & 3) * 8;
      uint4 kv = make_uint4(0u, 0u, 0u, 0u);
      uint4 vv = make_uint4(0u, 0u, 0u, 0u);
      if (kbase + row < LLEN) {
        kv = *(const uint4*)(Kmat + (base + kbase + row) * E_DIM + c0 + cv);
        vv = *(const uint4*)(Vmat + (base + kbase + row) * E_DIM + c0 + cv);
      }
      *(uint4*)&Kt[row][cv] = kv;
      union { uint4 u; u16b hcv[8]; } vu; vu.u = vv;
#pragma unroll
      for (int j = 0; j < 8; ++j) Vt[cv + j][row] = vu.hcv[j];
      if (kbase + 32 + row < LLEN)                   // gfx1250 global_prefetch
        __builtin_prefetch(Kmat + (base + kbase + 32 + row) * E_DIM + c0, 0, 1);
    }
    // fetch this tile's bias values (unconditional, aligned b128 loads)
    union { uint4 u[4]; u16b hb[32]; } bias;
    {
      const uint4* bp = (const uint4*)(brow + kbase);
#pragma unroll
      for (int i = 0; i < 4; ++i) bias.u[i] = bp[i];
    }
    __syncthreads();
    { // S = Q @ K^T  (reduction over d_head=32, one WMMA k-step)
      v8f s[2][2] = {};
      v16bf qa0 = frag_run8(&Qs[wave][r][0], hh);
      v16bf qa1 = frag_run8(&Qs[wave][16 + r][0], hh);
      v16bf kb0 = frag_run16(&Kt[r][0], hh);
      v16bf kb1 = frag_run16(&Kt[16 + r][0], hh);
      s[0][0] = __builtin_amdgcn_wmma_f32_16x16x32_bf16(false, qa0, false, kb0, (short)0, s[0][0], false, false);
      s[0][1] = __builtin_amdgcn_wmma_f32_16x16x32_bf16(false, qa0, false, kb1, (short)0, s[0][1], false, false);
      s[1][0] = __builtin_amdgcn_wmma_f32_16x16x32_bf16(false, qa1, false, kb0, (short)0, s[1][0], false, false);
      s[1][1] = __builtin_amdgcn_wmma_f32_16x16x32_bf16(false, qa1, false, kb1, (short)0, s[1][1], false, false);
#pragma unroll
      for (int mi = 0; mi < 2; ++mi)
#pragma unroll
        for (int ni = 0; ni < 2; ++ni)
#pragma unroll
          for (int i = 0; i < 8; ++i)
            Sls[wave][16 * mi + 8 * hh + i][16 * ni + r] = s[mi][ni][i];
    }
    __syncthreads();
    { // online softmax; lane = q row
      const float* srow_ = &Sls[wave][lane][0];
      float sv[32];
      float smax = -INFINITY;
#pragma unroll
      for (int j = 0; j < 32; ++j) {
        int kg = kbase + j;
        float x;
        if (kg < LLEN) {
          x = srow_[j] * scl + bsc * bf2f(bias.hb[j]);
        } else {
          x = -INFINITY;
        }
        sv[j] = x;
        smax = fmaxf(smax, x);
      }
      float newm = fmaxf(m_i, smax);
      float scale, psum = 0.f;
      if (newm == -INFINITY) {
        scale = 1.f;
#pragma unroll
        for (int j = 0; j < 32; ++j) Pls[wave][lane][j] = 0;
      } else {
        scale = (m_i == -INFINITY) ? 0.f : __expf(m_i - newm);
#pragma unroll
        for (int j = 0; j < 32; ++j) {
          float p = (sv[j] == -INFINITY) ? 0.f : __expf(sv[j] - newm);
          psum += p;
          Pls[wave][lane][j] = f2bf(p);
        }
      }
      l_i = l_i * scale + psum;
      m_i = newm;
      rowv[wave][lane] = scale;
    }
    __syncthreads();
    { // rescale O, then O += P @ V
#pragma unroll
      for (int mi = 0; mi < 2; ++mi)
#pragma unroll
        for (int i = 0; i < 8; ++i) {
          float sc2 = rowv[wave][16 * mi + 8 * hh + i];
          o[mi][0][i] *= sc2;
          o[mi][1][i] *= sc2;
        }
      v16bf pa0 = frag_run8(&Pls[wave][r][0], hh);
      v16bf pa1 = frag_run8(&Pls[wave][16 + r][0], hh);
      v16bf vb0 = frag_run16(&Vt[r][0], hh);
      v16bf vb1 = frag_run16(&Vt[16 + r][0], hh);
      o[0][0] = __builtin_amdgcn_wmma_f32_16x16x32_bf16(false, pa0, false, vb0, (short)0, o[0][0], false, false);
      o[0][1] = __builtin_amdgcn_wmma_f32_16x16x32_bf16(false, pa0, false, vb1, (short)0, o[0][1], false, false);
      o[1][0] = __builtin_amdgcn_wmma_f32_16x16x32_bf16(false, pa1, false, vb0, (short)0, o[1][0], false, false);
      o[1][1] = __builtin_amdgcn_wmma_f32_16x16x32_bf16(false, pa1, false, vb1, (short)0, o[1][1], false, false);
    }
  }
  __syncthreads();
  rowv[wave][lane] = l_i;
  __syncthreads();
#pragma unroll
  for (int mi = 0; mi < 2; ++mi)
#pragma unroll
    for (int ni = 0; ni < 2; ++ni)
#pragma unroll
      for (int i = 0; i < 8; ++i) {
        int row = 16 * mi + 8 * hh + i;
        int qg = qb + row;
        if (qg < LLEN) {
          float l = rowv[wave][row];
          float inv = (l > 0.f) ? 1.f / l : 0.f;
          Out[(base + qg) * E_DIM + c0 + 16 * ni + r] = f2bf(o[mi][ni][i] * inv);
        }
      }
}

// ---------------------------------------------------------------------------
// Final LN (special rows only) + risk MLP head. One block per batch element.
// ---------------------------------------------------------------------------
__global__ __launch_bounds__(256) void risk_head(
    const float* __restrict__ tokens, const float* __restrict__ fg,
    const float* __restrict__ fb, const float* __restrict__ w1,
    const float* __restrict__ b1, const float* __restrict__ lng,
    const float* __restrict__ lnb, const float* __restrict__ w2,
    const float* __restrict__ b2, const float* __restrict__ w3,
    const float* __restrict__ b3, float* __restrict__ out) {
  __shared__ float scr[8];
  __shared__ float feat[NSPEC * E_DIM];
  __shared__ float h1[E_DIM];
  __shared__ float h2[E_DIM / 2];
  const int b = blockIdx.x, e = threadIdx.x;
  for (int s0 = 0; s0 < NSPEC; ++s0) {
    float x = tokens[((size_t)b * LLEN + s0) * E_DIM + e];
    float s = block_sum256(x, scr), s2 = block_sum256(x * x, scr);
    float mu = s / E_DIM, var = s2 / E_DIM - mu * mu;
    feat[s0 * E_DIM + e] = (x - mu) * rsqrtf(var + LNEPS) * fg[e] + fb[e];
  }
  __syncthreads();
  float t = b1[e];
  for (int i = 0; i < NSPEC * E_DIM; ++i) t += feat[i] * w1[(size_t)i * E_DIM + e];
  float s = block_sum256(t, scr), s2 = block_sum256(t * t, scr);
  float mu = s / E_DIM, var = s2 / E_DIM - mu * mu;
  h1[e] = fmaxf((t - mu) * rsqrtf(var + LNEPS) * lng[e] + lnb[e], 0.f);
  __syncthreads();
  if (e < E_DIM / 2) {
    float t2 = b2[e];
    for (int i = 0; i < E_DIM; ++i) t2 += h1[i] * w2[i * (E_DIM / 2) + e];
    h2[e] = fmaxf(t2, 0.f);
  }
  __syncthreads();
  float v = (e < E_DIM / 2) ? h2[e] * w3[e] : 0.f;
  float tot = block_sum256(v, scr);
  if (e == 0) out[b] = tot + b3[0];
}

// ---------------------------------------------------------------------------
// Host launch. Input flattening: setup_inputs() insertion order, params
// pytree depth-first in insertion order (w before b) -> 96 leaves.
// ---------------------------------------------------------------------------
extern "C" void kernel_launch(void* const* d_in, const int* in_sizes, int n_in,
                              void* d_out, int out_size, void* d_ws, size_t ws_size,
                              hipStream_t stream) {
  (void)in_sizes; (void)n_in; (void)out_size; (void)ws_size;
  const float* protein  = (const float*)d_in[0];
  const float* clinical = (const float*)d_in[1];
  const float* genomic  = (const float*)d_in[2];
  const float* Kdiff    = (const float*)d_in[3];
  const float* PEm      = (const float*)d_in[4];
  const float* vp_w     = (const float*)d_in[5];
  const float* vp_b     = (const float*)d_in[6];
  const float* prot_emb = (const float*)d_in[7];
  const float* pe_w     = (const float*)d_in[8];
  const float* pe_b     = (const float*)d_in[9];
  const float* cls_tok  = (const float*)d_in[10];
  const float* clin_tok = (const float*)d_in[11];
  const float* gen_tok  = (const float*)d_in[12];
  const float* bias_sc  = (const float*)d_in[13];
  const float* cp_w = (const float*)d_in[14];
  const float* cp_b = (const float*)d_in[15];
  const float* cp_g = (const float*)d_in[16];
  const float* cp_e = (const float*)d_in[17];
  const float* gp_w = (const float*)d_in[18];
  const float* gp_b = (const float*)d_in[19];
  const float* gp_g = (const float*)d_in[20];
  const float* gp_e = (const float*)d_in[21];
  const float* Lw[NLAY][6];   // q,k,v,o,ffn1,ffn2 weights
  const float* Lb[NLAY][6];   // matching biases
  const float* Lln[NLAY][4];  // ln1_g, ln1_b, ln2_g, ln2_b
  for (int l = 0; l < NLAY; ++l) {
    int base = 22 + 16 * l;
    Lw[l][0] = (const float*)d_in[base + 0];  Lb[l][0] = (const float*)d_in[base + 1];
    Lw[l][1] = (const float*)d_in[base + 2];  Lb[l][1] = (const float*)d_in[base + 3];
    Lw[l][2] = (const float*)d_in[base + 4];  Lb[l][2] = (const float*)d_in[base + 5];
    Lw[l][3] = (const float*)d_in[base + 6];  Lb[l][3] = (const float*)d_in[base + 7];
    Lw[l][4] = (const float*)d_in[base + 8];  Lb[l][4] = (const float*)d_in[base + 9];
    Lw[l][5] = (const float*)d_in[base + 10]; Lb[l][5] = (const float*)d_in[base + 11];
    Lln[l][0] = (const float*)d_in[base + 12]; Lln[l][1] = (const float*)d_in[base + 13];
    Lln[l][2] = (const float*)d_in[base + 14]; Lln[l][3] = (const float*)d_in[base + 15];
  }
  const float* final_g = (const float*)d_in[86];
  const float* final_b = (const float*)d_in[87];
  const float* r_w1 = (const float*)d_in[88];
  const float* r_b1 = (const float*)d_in[89];
  const float* r_lg = (const float*)d_in[90];
  const float* r_lb = (const float*)d_in[91];
  const float* r_w2 = (const float*)d_in[92];
  const float* r_b2 = (const float*)d_in[93];
  const float* r_w3 = (const float*)d_in[94];
  const float* r_b3 = (const float*)d_in[95];

  // Workspace bump allocator (total ~62 MB).
  char* ws = (char*)d_ws;
  size_t off = 0;
  auto alloc = [&](size_t bytes) -> void* {
    void* p = ws + off;
    off = (off + bytes + 255) & ~(size_t)255;
    return p;
  };
  float* tokens = (float*)alloc((size_t)MROW * E_DIM * sizeof(float));
  u16b* hbuf = (u16b*)alloc((size_t)MROW * E_DIM * 2);
  u16b* qbuf = (u16b*)alloc((size_t)MROW * E_DIM * 2);
  u16b* kbuf = (u16b*)alloc((size_t)MROW * E_DIM * 2);
  u16b* vbuf = (u16b*)alloc((size_t)MROW * E_DIM * 2);
  u16b* abuf = (u16b*)alloc((size_t)MROW * E_DIM * 2);
  u16b* fbuf = (u16b*)alloc((size_t)MROW * FFN_D * 2);
  u16b* biasp = (u16b*)alloc((size_t)LLEN * BSTRIDE * 2);
  u16b* WT[NLAY][6];
  for (int l = 0; l < NLAY; ++l) {
    for (int j = 0; j < 4; ++j) WT[l][j] = (u16b*)alloc((size_t)E_DIM * E_DIM * 2);
    WT[l][4] = (u16b*)alloc((size_t)E_DIM * FFN_D * 2);
    WT[l][5] = (u16b*)alloc((size_t)FFN_D * E_DIM * 2);
  }

  // 1. one-time prep: weights -> bf16 transposed; padded bf16 bias matrix
  for (int l = 0; l < NLAY; ++l) {
    for (int j = 0; j < 4; ++j)
      weight_transpose_bf16<<<(E_DIM * E_DIM + 255) / 256, 256, 0, stream>>>(Lw[l][j], WT[l][j], E_DIM, E_DIM);
    weight_transpose_bf16<<<(E_DIM * FFN_D + 255) / 256, 256, 0, stream>>>(Lw[l][4], WT[l][4], E_DIM, FFN_D);
    weight_transpose_bf16<<<(E_DIM * FFN_D + 255) / 256, 256, 0, stream>>>(Lw[l][5], WT[l][5], FFN_D, E_DIM);
  }
  build_bias_pad<<<(int)(((size_t)LLEN * BSTRIDE + 255) / 256), 256, 0, stream>>>(Kdiff, biasp);

  // 2. token construction
  build_prot_tokens<<<dim3(NPROT, NBATCH), 256, 0, stream>>>(
      protein, PEm, vp_w, vp_b, pe_w, pe_b, prot_emb, tokens);
  build_special_tokens<<<NBATCH, 256, 0, stream>>>(
      clinical, genomic, cp_w, cp_b, cp_g, cp_e, gp_w, gp_b, gp_g, gp_e,
      cls_tok, clin_tok, gen_tok, tokens);

  // 3. transformer layers
  const int mt = (MROW + 63) / 64;  // 129
  for (int l = 0; l < NLAY; ++l) {
    ln_to_bf16<<<MROW, 256, 0, stream>>>(tokens, Lln[l][0], Lln[l][1], hbuf);
    gemm_bf16_wmma<0><<<dim3(E_DIM / 64, mt), 128, 0, stream>>>(hbuf, WT[l][0], Lb[l][0], qbuf, MROW, E_DIM, E_DIM);
    gemm_bf16_wmma<0><<<dim3(E_DIM / 64, mt), 128, 0, stream>>>(hbuf, WT[l][1], Lb[l][1], kbuf, MROW, E_DIM, E_DIM);
    gemm_bf16_wmma<0><<<dim3(E_DIM / 64, mt), 128, 0, stream>>>(hbuf, WT[l][2], Lb[l][2], vbuf, MROW, E_DIM, E_DIM);
    attn_flash_wmma<<<dim3((LLEN + 127) / 128, N_HEAD, NBATCH), 128, 0, stream>>>(
        qbuf, kbuf, vbuf, biasp, bias_sc, abuf);
    gemm_bf16_wmma<2><<<dim3(E_DIM / 64, mt), 128, 0, stream>>>(abuf, WT[l][3], Lb[l][3], tokens, MROW, E_DIM, E_DIM);
    ln_to_bf16<<<MROW, 256, 0, stream>>>(tokens, Lln[l][2], Lln[l][3], hbuf);
    gemm_bf16_wmma<1><<<dim3(FFN_D / 64, mt), 128, 0, stream>>>(hbuf, WT[l][4], Lb[l][4], fbuf, MROW, FFN_D, E_DIM);
    gemm_bf16_wmma<2><<<dim3(E_DIM / 64, mt), 128, 0, stream>>>(fbuf, WT[l][5], Lb[l][5], tokens, MROW, E_DIM, FFN_D);
  }

  // 4. final LN on special rows + risk head
  risk_head<<<NBATCH, 256, 0, stream>>>(tokens, final_g, final_b, r_w1, r_b1,
                                        r_lg, r_lb, r_w2, r_b2, r_w3, r_b3,
                                        (float*)d_out);
}
